// LocalSelfAttention1D_54589034332796
// MI455X (gfx1250) — matrix operand order repaired and checked by reading, not compiled, not run on hardware
//
#include <hip/hip_runtime.h>
#include <hip/hip_bf16.h>
#include <math.h>

#define BATCH   4
#define DIM     256
#define NHEADS  8
#define HDIM    32
#define LLEN    8192
#define WIN     256
#define STRIDE_ 192
#define NWIN    43      // ceil(8192/192)
#define HIDDEN_ 1024

typedef __bf16 bf16_t;
typedef __attribute__((ext_vector_type(16))) __bf16 v16bf;
typedef __attribute__((ext_vector_type(8)))  __bf16 v8bf;
typedef __attribute__((ext_vector_type(8)))  float  v8f;

__device__ __forceinline__ v8f zero8() {
  v8f z = {0.f,0.f,0.f,0.f,0.f,0.f,0.f,0.f};
  return z;
}

__device__ __forceinline__ v8f wmma_bf16(v16bf a, v16bf b, v8f c) {
  // D = A(16x32 bf16) * B(32x16 bf16) + C(16x16 f32)
  return __builtin_amdgcn_wmma_f32_16x16x32_bf16(false, a, false, b, (short)0, c, false, false);
}

// ---- fragment loaders (contiguous k order per lane, not the ISA table order; wave32) ----
// A (16x32 MxK): lane m = lane&15, half = lane>>4.
//   half=0: K=0..15, half=1: K=16..31 (contiguous per lane)
__device__ __forceinline__ v16bf load_a_frag_ptr(const bf16_t* rowp, int half) {
  union { v16bf v; v8bf h[2]; } u;
  u.h[0] = *(const v8bf*)(rowp + 16*half);
  u.h[1] = *(const v8bf*)(rowp + 16*half + 8);
  return u.v;
}
__device__ __forceinline__ v16bf load_a_frag(const bf16_t* p, int ld) {
  int lane = threadIdx.x & 31;
  return load_a_frag_ptr(p + (size_t)(lane & 15)*ld, lane >> 4);
}
// B (32x16 KxN), source "n-major": element (k,n) at p[n*ld + k].
// lanes 0-15 hold K=0..15, lanes 16-31 hold K=16..31 (contiguous per lane)
__device__ __forceinline__ v16bf load_b_frag(const bf16_t* p, int ld) {
  int lane = threadIdx.x & 31;
  return *(const v16bf*)(p + (size_t)(lane & 15)*ld + 16*(lane >> 4));
}

__device__ __forceinline__ float rowmax16(float v) {
#pragma unroll
  for (int m = 1; m < 16; m <<= 1) v = fmaxf(v, __shfl_xor(v, m, 32));
  return v;
}
__device__ __forceinline__ float rowsum16(float v) {
#pragma unroll
  for (int m = 1; m < 16; m <<= 1) v += __shfl_xor(v, m, 32);
  return v;
}

// ---- shared GEMM core: one wave computes a 32x32 output tile ----
// Wm: weight rows (m_base..m_base+31), row-major, ld = K
// Xn: activations at (n_base..n_base+31), position-major, ld = K
template<int K>
__device__ __forceinline__ void gemm_tile(const bf16_t* __restrict__ Wm,
                                          const bf16_t* __restrict__ Xn,
                                          v8f acc[2][2]) {
#pragma unroll 4
  for (int k = 0; k < K; k += 32) {
    v16bf a0 = load_a_frag(Wm + k, K);
    v16bf a1 = load_a_frag(Wm + (size_t)16*K + k, K);
    v16bf b0 = load_b_frag(Xn + k, K);
    v16bf b1 = load_b_frag(Xn + (size_t)16*K + k, K);
    acc[0][0] = wmma_bf16(a0, b0, acc[0][0]);
    acc[0][1] = wmma_bf16(a0, b1, acc[0][1]);
    acc[1][0] = wmma_bf16(a1, b0, acc[1][0]);
    acc[1][1] = wmma_bf16(a1, b1, acc[1][1]);
  }
}

// ======================= prep kernels =======================
__global__ void lsa_f2bf_kernel(const float* __restrict__ src, bf16_t* __restrict__ dst, int n) {
  int i = blockIdx.x * 256 + threadIdx.x;
  if (i < n) dst[i] = (bf16_t)src[i];
}

// xp_t[(b*L+l)*C + c] = bf16(x[b][c][l] + pos[c])
__global__ void lsa_build_xp_kernel(const float* __restrict__ x, const float* __restrict__ pos,
                                    bf16_t* __restrict__ xp_t) {
  size_t i = (size_t)blockIdx.x * 256 + threadIdx.x;
  int c = (int)(i % DIM);
  size_t bl = i / DIM;
  int l = (int)(bl % LLEN);
  int b = (int)(bl / LLEN);
  float v = x[((size_t)b*DIM + c)*LLEN + l] + pos[c];
  xp_t[i] = (bf16_t)v;
}

// ======================= qkv GEMM =======================
// out channels o: t=o>>8 (q/k/v), h=(o>>5)&7, d=o&31 -> qkvh[t][(b*8+h)*L + l][d]
__global__ void __launch_bounds__(256)
lsa_qkv_gemm_kernel(const bf16_t* __restrict__ Wq, const float* __restrict__ bias,
                    const bf16_t* __restrict__ xp, bf16_t* __restrict__ qkvh) {
  int b = blockIdx.z;
  int wave = threadIdx.x >> 5, lane = threadIdx.x & 31;
  int m0 = blockIdx.y*64 + (wave >> 2)*32;
  int n0 = blockIdx.x*128 + (wave & 3)*32;
  const bf16_t* Wm = Wq + (size_t)m0*DIM;
  const bf16_t* Xn = xp + ((size_t)b*LLEN + n0)*DIM;
  v8f acc[2][2] = { {zero8(), zero8()}, {zero8(), zero8()} };
  gemm_tile<DIM>(Wm, Xn, acc);
  int half = lane >> 4, nn = lane & 15;
  const size_t TSZ = (size_t)BATCH*NHEADS*LLEN*HDIM;
#pragma unroll
  for (int mt = 0; mt < 2; ++mt)
#pragma unroll
    for (int nt = 0; nt < 2; ++nt) {
      int n  = n0 + nt*16 + nn;
      int mb = m0 + mt*16 + 8*half;
      int t = mb >> 8, h = (mb >> 5) & 7, d0 = mb & 31;
      bf16_t* dst = qkvh + (size_t)t*TSZ + (((size_t)(b*NHEADS + h)*LLEN + n)*HDIM) + d0;
#pragma unroll
      for (int r = 0; r < 8; ++r)
        dst[r] = (bf16_t)(acc[mt][nt][r] + bias[mb + r]);
    }
}

// ======================= windowed flash attention =======================
__global__ void __launch_bounds__(256)
lsa_attn_kernel(const bf16_t* __restrict__ qkvh, bf16_t* __restrict__ win_out) {
  const size_t TSZ = (size_t)BATCH*NHEADS*LLEN*HDIM;
  int w  = blockIdx.x;     // window index
  int bh = blockIdx.y;     // b*NHEADS + h
  int start = w * STRIDE_;
  const bf16_t* Q  = qkvh + (size_t)bh*LLEN*HDIM;
  const bf16_t* Kp = qkvh + TSZ   + (size_t)bh*LLEN*HDIM;
  const bf16_t* Vp = qkvh + 2*TSZ + (size_t)bh*LLEN*HDIM;

  __shared__ __attribute__((aligned(64))) bf16_t Kt[32*32];      // (key, d)
  __shared__ __attribute__((aligned(64))) bf16_t Vt[32*32];      // transposed (d, key)
  __shared__ __attribute__((aligned(64))) bf16_t Pst[8][16*32];  // per-wave P staging

  int wave = threadIdx.x >> 5, lane = threadIdx.x & 31;
  int half = lane >> 4, nn = lane & 15;

  // Q fragments for this wave's 2 query row-tiles (reused across key tiles)
  v16bf qf[2];
#pragma unroll
  for (int t = 0; t < 2; ++t) {
    int ql = wave*32 + t*16 + nn;
    int qg = start + ql; if (qg > LLEN-1) qg = LLEN-1;
    qf[t] = load_a_frag_ptr(Q + (size_t)qg*HDIM, half);
  }

  v8f acc[2][2] = { {zero8(), zero8()}, {zero8(), zero8()} };
  float mrow[2][8], srow[2][8];
#pragma unroll
  for (int t = 0; t < 2; ++t)
#pragma unroll
    for (int r = 0; r < 8; ++r) { mrow[t][r] = -1e30f; srow[t][r] = 0.f; }

  const float scale = 0.17677669529663687f;  // 1/sqrt(32)

  for (int kt = 0; kt < WIN/32; ++kt) {
    __syncthreads();
    { // cooperative stage of 32x32 K and V tiles
      int tid = threadIdx.x;
      int key_l = tid >> 3;
      int d0 = (tid & 7) * 4;
      int kg = start + kt*32 + key_l; if (kg > LLEN-1) kg = LLEN-1;
      const bf16_t* ksrc = Kp + (size_t)kg*HDIM + d0;
      const bf16_t* vsrc = Vp + (size_t)kg*HDIM + d0;
#pragma unroll
      for (int j = 0; j < 4; ++j) {
        Kt[key_l*32 + d0 + j] = ksrc[j];
        Vt[(d0 + j)*32 + key_l] = vsrc[j];
      }
    }
    __syncthreads();

#pragma unroll
    for (int t = 0; t < 2; ++t) {
      // scores: 16 queries x 32 keys (two 16-wide N tiles)
      v8f s0 = wmma_bf16(qf[t], load_b_frag(Kt, 32),         zero8());
      v8f s1 = wmma_bf16(qf[t], load_b_frag(Kt + 16*32, 32), zero8());
      int k0g = start + kt*32 + nn;
      bool v0 = (k0g      < LLEN);
      bool v1 = (k0g + 16 < LLEN);
      float p0[8], p1[8], mnew[8];
#pragma unroll
      for (int r = 0; r < 8; ++r) {
        float a = v0 ? s0[r]*scale : -1e30f;
        float c = v1 ? s1[r]*scale : -1e30f;
        s0[r] = a; s1[r] = c;
        float tmax = rowmax16(fmaxf(a, c));
        mnew[r] = fmaxf(mrow[t][r], tmax);
      }
#pragma unroll
      for (int r = 0; r < 8; ++r) {
        float alpha = __expf(mrow[t][r] - mnew[r]);
        mrow[t][r] = mnew[r];
        p0[r] = __expf(s0[r] - mnew[r]);
        p1[r] = __expf(s1[r] - mnew[r]);
        float rs = rowsum16(p0[r] + p1[r]);
        srow[t][r] = srow[t][r]*alpha + rs;
        acc[t][0][r] *= alpha;
        acc[t][1][r] *= alpha;
      }
      // re-fragment P (D-layout -> A-layout) through per-wave LDS
      bf16_t* ps = Pst[wave];
#pragma unroll
      for (int r = 0; r < 8; ++r) {
        int row = r + 8*half;
        ps[row*32 + nn]      = (bf16_t)p0[r];
        ps[row*32 + 16 + nn] = (bf16_t)p1[r];
      }
      v16bf pf = load_a_frag(ps, 32);
      acc[t][0] = wmma_bf16(pf, load_b_frag(Vt, 32),         acc[t][0]);
      acc[t][1] = wmma_bf16(pf, load_b_frag(Vt + 16*32, 32), acc[t][1]);
    }
  }

  // epilogue: normalize by row sum, zero invalid queries, store window output
  bf16_t* wout = win_out + ((size_t)bh*NWIN + w)*WIN*HDIM;
#pragma unroll
  for (int t = 0; t < 2; ++t)
#pragma unroll
    for (int dt = 0; dt < 2; ++dt)
#pragma unroll
      for (int r = 0; r < 8; ++r) {
        int ql = wave*32 + t*16 + r + 8*half;
        bool qv = (start + ql) < LLEN;
        float ov = qv ? acc[t][dt][r] / srow[t][r] : 0.f;
        wout[(size_t)ql*HDIM + dt*16 + nn] = (bf16_t)ov;
      }
}

// ======================= overlap gather / average =======================
__global__ void lsa_gather_kernel(const bf16_t* __restrict__ win_out, bf16_t* __restrict__ attn_t) {
  size_t i = (size_t)blockIdx.x * 256 + threadIdx.x;   // over (b, l, c), c fastest
  int c = (int)(i % DIM);
  int d = c & 31, h = c >> 5;
  size_t bl = i / DIM;
  int l = (int)(bl % LLEN);
  int b = (int)(bl / LLEN);
  int whi = l / STRIDE_;
  float s = 0.f; int cnt = 0;
#pragma unroll
  for (int k = 0; k < 2; ++k) {
    int wi = whi - k;
    int offp = l - wi*STRIDE_;
    if (wi >= 0 && wi < NWIN && offp < WIN) {
      s += (float)win_out[(((size_t)(b*NHEADS + h)*NWIN + wi)*WIN + offp)*HDIM + d];
      ++cnt;
    }
  }
  attn_t[i] = (bf16_t)(s / (float)cnt);
}

// ======================= output projection + residual =======================
__global__ void __launch_bounds__(256)
lsa_outproj_gemm_kernel(const bf16_t* __restrict__ Wo, const float* __restrict__ bias,
                        const bf16_t* __restrict__ attn_t, const float* __restrict__ x,
                        float* __restrict__ y_t) {
  int b = blockIdx.z;
  int wave = threadIdx.x >> 5, lane = threadIdx.x & 31;
  int m0 = blockIdx.y*64 + (wave >> 2)*32;
  int n0 = blockIdx.x*128 + (wave & 3)*32;
  v8f acc[2][2] = { {zero8(), zero8()}, {zero8(), zero8()} };
  gemm_tile<DIM>(Wo + (size_t)m0*DIM, attn_t + ((size_t)b*LLEN + n0)*DIM, acc);
  int half = lane >> 4, nn = lane & 15;
#pragma unroll
  for (int mt = 0; mt < 2; ++mt)
#pragma unroll
    for (int nt = 0; nt < 2; ++nt) {
      int n  = n0 + nt*16 + nn;
      int mb = m0 + mt*16 + 8*half;
      float* dst = y_t + ((size_t)b*LLEN + n)*DIM + mb;
#pragma unroll
      for (int r = 0; r < 8; ++r) {
        int m = mb + r;
        dst[r] = acc[mt][nt][r] + bias[m] + x[((size_t)b*DIM + m)*LLEN + n];
      }
    }
}

// ======================= instance-norm stats (per b,channel over L) =======================
__global__ void __launch_bounds__(256)
lsa_stats_kernel(const float* __restrict__ src, float* __restrict__ stats) {
  int bc = blockIdx.x;               // b*DIM + c
  int b = bc / DIM, c = bc % DIM;
  const float* p = src + (size_t)b*LLEN*DIM + c;
  float s = 0.f, s2 = 0.f;
  for (int l = threadIdx.x; l < LLEN; l += 256) {
    float v = p[(size_t)l*DIM];
    s += v; s2 += v*v;
  }
  __shared__ float sh[256], sh2[256];
  sh[threadIdx.x] = s; sh2[threadIdx.x] = s2;
  __syncthreads();
  for (int o = 128; o > 0; o >>= 1) {
    if (threadIdx.x < o) { sh[threadIdx.x] += sh[threadIdx.x + o]; sh2[threadIdx.x] += sh2[threadIdx.x + o]; }
    __syncthreads();
  }
  if (threadIdx.x == 0) {
    float mean = sh[0] / (float)LLEN;
    float var  = sh2[0] / (float)LLEN - mean*mean;
    stats[2*bc]   = mean;
    stats[2*bc+1] = rsqrtf(var + 1e-5f);
  }
}

__global__ void lsa_norm_bf16_kernel(const float* __restrict__ y, const float* __restrict__ stats,
                                     bf16_t* __restrict__ out) {
  size_t i = (size_t)blockIdx.x * 256 + threadIdx.x;
  int c = (int)(i % DIM);
  int b = (int)((i / DIM) / LLEN);
  int bc = b*DIM + c;
  out[i] = (bf16_t)((y[i] - stats[2*bc]) * stats[2*bc+1]);
}

// ======================= MLP GEMMs =======================
__global__ void __launch_bounds__(256)
lsa_mlp1_gemm_kernel(const bf16_t* __restrict__ W1, const float* __restrict__ bias,
                     const bf16_t* __restrict__ y_n, bf16_t* __restrict__ h_t) {
  int b = blockIdx.z;
  int wave = threadIdx.x >> 5, lane = threadIdx.x & 31;
  int m0 = blockIdx.y*64 + (wave >> 2)*32;
  int n0 = blockIdx.x*128 + (wave & 3)*32;
  v8f acc[2][2] = { {zero8(), zero8()}, {zero8(), zero8()} };
  gemm_tile<DIM>(W1 + (size_t)m0*DIM, y_n + ((size_t)b*LLEN + n0)*DIM, acc);
  int half = lane >> 4, nn = lane & 15;
#pragma unroll
  for (int mt = 0; mt < 2; ++mt)
#pragma unroll
    for (int nt = 0; nt < 2; ++nt) {
      int n  = n0 + nt*16 + nn;
      int mb = m0 + mt*16 + 8*half;
      bf16_t* dst = h_t + ((size_t)b*LLEN + n)*HIDDEN_ + mb;
#pragma unroll
      for (int r = 0; r < 8; ++r) {
        float v = acc[mt][nt][r] + bias[mb + r];
        float g = 0.5f * v * (1.f + erff(v * 0.70710678118654752f));  // exact gelu
        dst[r] = (bf16_t)g;
      }
    }
}

__global__ void __launch_bounds__(256)
lsa_mlp2_gemm_kernel(const bf16_t* __restrict__ W2, const float* __restrict__ bias,
                     const bf16_t* __restrict__ h_t, float* __restrict__ m_t) {
  int b = blockIdx.z;
  int wave = threadIdx.x >> 5, lane = threadIdx.x & 31;
  int m0 = blockIdx.y*64 + (wave >> 2)*32;
  int n0 = blockIdx.x*128 + (wave & 3)*32;
  v8f acc[2][2] = { {zero8(), zero8()}, {zero8(), zero8()} };
  gemm_tile<HIDDEN_>(W2 + (size_t)m0*HIDDEN_, h_t + ((size_t)b*LLEN + n0)*HIDDEN_, acc);
  int half = lane >> 4, nn = lane & 15;
#pragma unroll
  for (int mt = 0; mt < 2; ++mt)
#pragma unroll
    for (int nt = 0; nt < 2; ++nt) {
      int n  = n0 + nt*16 + nn;
      int mb = m0 + mt*16 + 8*half;
      float* dst = m_t + ((size_t)b*LLEN + n)*DIM + mb;
#pragma unroll
      for (int r = 0; r < 8; ++r)
        dst[r] = acc[mt][nt][r] + bias[mb + r];
    }
}

// ======================= final: norm(m) + norm(y), transpose to (b,c,l) =======================
__global__ void lsa_final_kernel(const float* __restrict__ m_t, const float* __restrict__ mstats,
                                 const float* __restrict__ y_t, const float* __restrict__ ystats,
                                 float* __restrict__ out) {
  size_t i = (size_t)blockIdx.x * 256 + threadIdx.x;   // over (b, c, l), l fastest
  int l = (int)(i % LLEN);
  size_t t = i / LLEN;
  int c = (int)(t % DIM);
  int b = (int)(t / DIM);
  size_t src = ((size_t)b*LLEN + l)*DIM + c;
  int bc = b*DIM + c;
  float mv = (m_t[src] - mstats[2*bc]) * mstats[2*bc+1];
  float yv = (y_t[src] - ystats[2*bc]) * ystats[2*bc+1];
  out[i] = mv + yv;
}

// ======================= host =======================
extern "C" void kernel_launch(void* const* d_in, const int* in_sizes, int n_in,
                              void* d_out, int out_size, void* d_ws, size_t ws_size,
                              hipStream_t stream) {
  const float* x      = (const float*)d_in[0];
  const float* pos    = (const float*)d_in[1];
  const float* w_qkv  = (const float*)d_in[2];
  const float* b_qkv  = (const float*)d_in[3];
  const float* w_out  = (const float*)d_in[4];
  const float* b_out  = (const float*)d_in[5];
  const float* w_mlp1 = (const float*)d_in[6];
  const float* b_mlp1 = (const float*)d_in[7];
  const float* w_mlp2 = (const float*)d_in[8];
  const float* b_mlp2 = (const float*)d_in[9];
  float* out = (float*)d_out;

  char* ws = (char*)d_ws;
  size_t off = 0;
  auto alloc = [&](size_t bytes) -> char* {
    char* p = ws + off;
    off += (bytes + 255) & ~(size_t)255;
    return p;
  };

  const size_t N_BLC  = (size_t)BATCH*LLEN*DIM;                 // 8388608
  const size_t N_QKVH = (size_t)BATCH*NHEADS*LLEN*HDIM;         // 8388608
  const size_t N_WOUT = (size_t)BATCH*NHEADS*NWIN*WIN*HDIM;     // 11272192
  const size_t N_H    = (size_t)BATCH*LLEN*HIDDEN_;             // 33554432

  bf16_t* wqkv_bf = (bf16_t*)alloc((size_t)768*256*2);
  bf16_t* wout_bf = (bf16_t*)alloc((size_t)256*256*2);
  bf16_t* w1_bf   = (bf16_t*)alloc((size_t)1024*256*2);
  bf16_t* w2_bf   = (bf16_t*)alloc((size_t)256*1024*2);
  bf16_t* xp_t    = (bf16_t*)alloc(N_BLC*2);
  bf16_t* qkvh    = (bf16_t*)alloc(3*N_QKVH*2);
  bf16_t* win_o   = (bf16_t*)alloc(N_WOUT*2);
  bf16_t* attn_t  = (bf16_t*)alloc(N_BLC*2);
  float*  y_t     = (float*) alloc(N_BLC*4);
  float*  ystats  = (float*) alloc((size_t)BATCH*DIM*2*4);
  bf16_t* y_n     = (bf16_t*)alloc(N_BLC*2);
  bf16_t* h_t     = (bf16_t*)alloc(N_H*2);
  float*  m_t     = (float*) alloc(N_BLC*4);
  float*  mstats  = (float*) alloc((size_t)BATCH*DIM*2*4);
  (void)ws_size; (void)in_sizes; (void)n_in; (void)out_size;

  // 1) weights -> bf16
  lsa_f2bf_kernel<<<(768*256+255)/256, 256, 0, stream>>>(w_qkv,  wqkv_bf, 768*256);
  lsa_f2bf_kernel<<<(256*256+255)/256, 256, 0, stream>>>(w_out,  wout_bf, 256*256);
  lsa_f2bf_kernel<<<(1024*256+255)/256,256, 0, stream>>>(w_mlp1, w1_bf,  1024*256);
  lsa_f2bf_kernel<<<(256*1024+255)/256,256, 0, stream>>>(w_mlp2, w2_bf,  256*1024);

  // 2) xp = x + pos, transposed to (b,l,c) bf16
  lsa_build_xp_kernel<<<(unsigned)(N_BLC/256), 256, 0, stream>>>(x, pos, xp_t);

  // 3) qkv projection (M=768, N=8192, K=256) -> head-major q/k/v
  lsa_qkv_gemm_kernel<<<dim3(LLEN/128, 768/64, BATCH), 256, 0, stream>>>(wqkv_bf, b_qkv, xp_t, qkvh);

  // 4) windowed flash attention
  lsa_attn_kernel<<<dim3(NWIN, BATCH*NHEADS), 256, 0, stream>>>(qkvh, win_o);

  // 5) gather overlapping windows, average by coverage
  lsa_gather_kernel<<<(unsigned)(N_BLC/256), 256, 0, stream>>>(win_o, attn_t);

  // 6) output projection + bias + residual
  lsa_outproj_gemm_kernel<<<dim3(LLEN/128, DIM/64, BATCH), 256, 0, stream>>>(wout_bf, b_out, attn_t, x, y_t);

  // 7) instance norm of y: stats + normalized bf16 copy
  lsa_stats_kernel<<<BATCH*DIM, 256, 0, stream>>>(y_t, ystats);
  lsa_norm_bf16_kernel<<<(unsigned)(N_BLC/256), 256, 0, stream>>>(y_t, ystats, y_n);

  // 8) MLP
  lsa_mlp1_gemm_kernel<<<dim3(LLEN/128, HIDDEN_/64, BATCH), 256, 0, stream>>>(w1_bf, b_mlp1, y_n, h_t);
  lsa_mlp2_gemm_kernel<<<dim3(LLEN/128, DIM/64, BATCH), 256, 0, stream>>>(w2_bf, b_mlp2, h_t, m_t);

  // 9) instance norm of m + add normalized y, transpose to (b,c,l)
  lsa_stats_kernel<<<BATCH*DIM, 256, 0, stream>>>(m_t, mstats);
  lsa_final_kernel<<<(unsigned)(N_BLC/256), 256, 0, stream>>>(m_t, mstats, y_t, ystats, out);
}
